// GraphSST2Net_9242769621975
// MI455X (gfx1250) — matrix operand order repaired and checked by reading, not compiled, not run on hardware
//
#include <hip/hip_runtime.h>
#include <hip/hip_bf16.h>

// ---------------------------------------------------------------------------
// GraphSST2 ARMA-GNN pipeline for MI455X (gfx1250, wave32, WMMA)
//
// N=100000 nodes, E=1600000 edges, F_IN=256, HID=128, G=512, C=2.
//
//   1. deg[col] += w (atomic) -> dinv = rsqrt(deg); norm = dinv[row]*w*dinv[col]
//   2. weights pre-packed to f16 in WMMA B-register layout (one-time, tiny)
//   3. conv0: H = X@Wi0, AGG0 = X@Wr0 + b0   (dual WMMA GEMM, 2 M-tiles/wave)
//      scatter: AGG0[col] += norm*H[row]; h = relu(AGG0)
//   4. conv1: same with Wi1/Wr1 on h -> AGG1
//   5. pool: gsum[batch] += relu(AGG1), gcnt[batch]++
//   6. MLP: hidden = relu((gsum/gcnt)@W1+b1) ; out = hidden@W2+b2
// ---------------------------------------------------------------------------

typedef __attribute__((ext_vector_type(16))) _Float16 v16h;
typedef __attribute__((ext_vector_type(8)))  float    v8f;

#define HID    128
#define NGRAPH 512
#define MLPH   256
#define NCLS   2

// ------------------------------ utility kernels ----------------------------

__global__ void zero_kernel(float* __restrict__ p, long n) {
    long i = (long)blockIdx.x * blockDim.x + threadIdx.x;
    if (i < n) p[i] = 0.0f;
}

__global__ void relu_kernel(float* __restrict__ p, long n) {
    long i = (long)blockIdx.x * blockDim.x + threadIdx.x;
    if (i < n) { float v = p[i]; p[i] = v > 0.0f ? v : 0.0f; }
}

// ------------------------------ edge norm ----------------------------------

__global__ void deg_kernel(const int* __restrict__ col,
                           const float* __restrict__ w,
                           float* __restrict__ deg, int E) {
    int e = blockIdx.x * blockDim.x + threadIdx.x;
    if (e < E) atomicAdd(&deg[col[e]], w[e]);
}

__global__ void dinv_kernel(float* __restrict__ deg, int n) {
    int i = blockIdx.x * blockDim.x + threadIdx.x;
    if (i < n) { float d = deg[i]; deg[i] = d > 0.0f ? __frsqrt_rn(d) : 0.0f; }
}

__global__ void norm_kernel(const int* __restrict__ row,
                            const int* __restrict__ col,
                            const float* __restrict__ w,
                            const float* __restrict__ dinv,
                            float* __restrict__ norm, int E) {
    int e = blockIdx.x * blockDim.x + threadIdx.x;
    if (e < E) norm[e] = dinv[row[e]] * w[e] * dinv[col[e]];
}

// --------------------- weight pre-pack into WMMA B layout ------------------
// W [K, HID] f32 row-major  ->  Bp f16, laid out so a wave handling column
// tile nt reads, per lane, ONE contiguous 32-byte v16h per K-chunk:
//   idx = c*4096 + nt*512 + lane*16 + i
//   Bp[idx] = W[(c*32 + (lane>=16 ? 8:0) + (i&7) + (i>=8 ? 16:0)) * HID + nt*16 + (lane&15)]
// (CDNA5 ISA 7.12.2: B 32x16 f16 — lane<16 col N=lane, halves 0-7 -> K+0..7, 8-15 -> K+16..23;
//  lane>=16 col N=lane-16, halves 0-7 -> K+8..15, 8-15 -> K+24..31.)

__global__ void pack_w_kernel(const float* __restrict__ W,
                              _Float16* __restrict__ Bp, int K) {
    int idx = blockIdx.x * blockDim.x + threadIdx.x;
    if (idx >= K * HID) return;
    int i    = idx & 15;
    int lane = (idx >> 4) & 31;
    int nt   = (idx >> 9) & 7;
    int c    = idx >> 12;
    int k = c * 32 + ((lane & 16) ? 8 : 0) + (i & 7) + ((i & 8) ? 16 : 0);
    int n = nt * 16 + (lane & 15);
    Bp[idx] = (_Float16)W[k * HID + n];
}

// --------------------- fused dual WMMA GEMM (f16 in, f32 acc) --------------
// H   = X @ Wi ; AGG = X @ Wr + bias     (both [N, 128])
// Block = (32, 8): 8 waves cover the 8 column tiles of a 32-row strip.
// Each wave computes TWO 16x16 output tiles (M and M+16) so every B
// fragment feeds two WMMAs. A fragments per ISA 7.12.2:
//   lanes 0-15 row M=lane,  halves 0-7 -> K+0..7,  8-15 -> K+16..23
//   lanes 16-31 row M=l-16, halves 0-7 -> K+8..15, 8-15 -> K+24..31

__global__ __launch_bounds__(256) void gemm_dual_wmma(
    const float* __restrict__ X,
    const _Float16* __restrict__ Bi, const _Float16* __restrict__ Br,
    const float* __restrict__ bias,
    float* __restrict__ H, float* __restrict__ AGG, int K) {

    const int lane = threadIdx.x;          // 0..31
    const int nt   = threadIdx.y;          // 0..7 column tile
    const int m0   = blockIdx.x * 32;      // 32-row strip
    const int n0   = nt * 16;
    const bool hi  = lane >= 16;
    const int  r   = lane & 15;

    v8f ci0 = {}, ci1 = {}, cr0 = {}, cr1 = {};

    const float* x0 = X + (size_t)(m0 + r) * K + (hi ? 8 : 0);
    const float* x1 = x0 + (size_t)16 * K;
    const _Float16* bip = Bi + ((size_t)nt * 512 + (size_t)lane * 16);
    const _Float16* brp = Br + ((size_t)nt * 512 + (size_t)lane * 16);

    for (int kc = 0; kc < K; kc += 32) {
        const float* p0 = x0 + kc;
        const float* p1 = x1 + kc;
        __builtin_prefetch(p0 + 32, 0, 1);             // global_prefetch_b8

        float t0[16], t1[16];
        *(float4*)&t0[0]  = *(const float4*)(p0);      // b128 loads
        *(float4*)&t0[4]  = *(const float4*)(p0 + 4);
        *(float4*)&t0[8]  = *(const float4*)(p0 + 16);
        *(float4*)&t0[12] = *(const float4*)(p0 + 20);
        *(float4*)&t1[0]  = *(const float4*)(p1);
        *(float4*)&t1[4]  = *(const float4*)(p1 + 4);
        *(float4*)&t1[8]  = *(const float4*)(p1 + 16);
        *(float4*)&t1[12] = *(const float4*)(p1 + 20);

        v16h a0, a1;
#pragma unroll
        for (int i = 0; i < 16; ++i) {                 // v_cvt_pk_f16_f32 pairs
            a0[i] = (_Float16)t0[i];
            a1[i] = (_Float16)t1[i];
        }

        const size_t coff = (size_t)(kc >> 5) * 4096;  // packed chunk stride
        v16h bi = *(const v16h*)(bip + coff);          // one 32B fragment/lane
        v16h br = *(const v16h*)(brp + coff);

        ci0 = __builtin_amdgcn_wmma_f32_16x16x32_f16(false, a0, false, bi,
                                                     (short)0, ci0, false, false);
        cr0 = __builtin_amdgcn_wmma_f32_16x16x32_f16(false, a0, false, br,
                                                     (short)0, cr0, false, false);
        ci1 = __builtin_amdgcn_wmma_f32_16x16x32_f16(false, a1, false, bi,
                                                     (short)0, ci1, false, false);
        cr1 = __builtin_amdgcn_wmma_f32_16x16x32_f16(false, a1, false, br,
                                                     (short)0, cr1, false, false);
    }

    // C/D layout: lane<16 vgpr v -> (mb+v, n0+lane); lane>=16 -> (mb+8+v, ...)
    const int n  = n0 + r;
    const float bv = bias[n];
    const int mb0 = m0 + (hi ? 8 : 0);
    const int mb1 = mb0 + 16;
#pragma unroll
    for (int v = 0; v < 8; ++v) {
        H  [(size_t)(mb0 + v) * HID + n] = ci0[v];
        AGG[(size_t)(mb0 + v) * HID + n] = cr0[v] + bv;
        H  [(size_t)(mb1 + v) * HID + n] = ci1[v];
        AGG[(size_t)(mb1 + v) * HID + n] = cr1[v] + bv;
    }
}

// --------------------------- edge scatter-add ------------------------------
// AGG[col[e], :] += norm[e] * H[row[e], :]
// 128 threads/block = 4 edges; each lane handles 4 features (float4 gather).

__global__ __launch_bounds__(128) void scatter_kernel(
    const int* __restrict__ row, const int* __restrict__ col,
    const float* __restrict__ norm, const float* __restrict__ H,
    float* __restrict__ AGG, int E) {

    int tid = threadIdx.x;
    int e = blockIdx.x * 4 + (tid >> 5);
    if (e >= E) return;
    int f = (tid & 31) * 4;
    int rs = row[e], cs = col[e];
    float w = norm[e];
    const float4 hv = *(const float4*)(H + (size_t)rs * HID + f);
    float* dst = AGG + (size_t)cs * HID + f;
    atomicAdd(dst + 0, w * hv.x);
    atomicAdd(dst + 1, w * hv.y);
    atomicAdd(dst + 2, w * hv.z);
    atomicAdd(dst + 3, w * hv.w);
}

// ------------------------------ mean pool ----------------------------------

__global__ void pool_kernel(const float* __restrict__ AGG1,
                            const int* __restrict__ batch,
                            float* __restrict__ gsum,
                            float* __restrict__ gcnt, int N) {
    long idx = (long)blockIdx.x * blockDim.x + threadIdx.x;
    int node = (int)(idx >> 7);            // /128
    int f    = (int)(idx & 127);
    if (node >= N) return;
    int g = batch[node];
    float v = AGG1[(size_t)node * HID + f];
    v = v > 0.0f ? v : 0.0f;               // final relu of conv1
    atomicAdd(&gsum[g * HID + f], v);
    if (f == 0) atomicAdd(&gcnt[g], 1.0f);
}

// ------------------------------ classifier MLP -----------------------------

__global__ void mlp_hidden_kernel(const float* __restrict__ gsum,
                                  const float* __restrict__ gcnt,
                                  const float* __restrict__ w1,
                                  const float* __restrict__ b1,
                                  float* __restrict__ hidden) {
    int idx = blockIdx.x * blockDim.x + threadIdx.x;   // G*MLPH
    if (idx >= NGRAPH * MLPH) return;
    int g = idx >> 8, j = idx & (MLPH - 1);
    float inv = 1.0f / fmaxf(gcnt[g], 1.0f);
    float acc = b1[j];
    for (int k = 0; k < HID; ++k)
        acc += (gsum[g * HID + k] * inv) * w1[k * MLPH + j];
    hidden[idx] = acc > 0.0f ? acc : 0.0f;
}

__global__ void mlp_out_kernel(const float* __restrict__ hidden,
                               const float* __restrict__ w2,
                               const float* __restrict__ b2,
                               float* __restrict__ out) {
    int idx = blockIdx.x * blockDim.x + threadIdx.x;   // G*NCLS
    if (idx >= NGRAPH * NCLS) return;
    int g = idx >> 1, c = idx & 1;
    float acc = b2[c];
    for (int k = 0; k < MLPH; ++k)
        acc += hidden[g * MLPH + k] * w2[k * NCLS + c];
    out[idx] = acc;
}

// ------------------------------- launcher ----------------------------------

extern "C" void kernel_launch(void* const* d_in, const int* in_sizes, int n_in,
                              void* d_out, int out_size, void* d_ws, size_t ws_size,
                              hipStream_t stream) {
    const float* x      = (const float*)d_in[0];
    const int*   eidx   = (const int*)  d_in[1];
    const float* eattr  = (const float*)d_in[2];
    const int*   batch  = (const int*)  d_in[3];
    const float* wi0    = (const float*)d_in[4];
    const float* wr0    = (const float*)d_in[5];
    const float* b0     = (const float*)d_in[6];
    const float* wi1    = (const float*)d_in[7];
    const float* wr1    = (const float*)d_in[8];
    const float* b1     = (const float*)d_in[9];
    const float* mw1    = (const float*)d_in[10];
    const float* mb1    = (const float*)d_in[11];
    const float* mw2    = (const float*)d_in[12];
    const float* mb2    = (const float*)d_in[13];
    float* out = (float*)d_out;

    const int N = in_sizes[0] / 256;      // 100000
    const int E = in_sizes[2];            // 1600000
    const int* row = eidx;
    const int* col = eidx + E;

    // workspace layout (floats, then f16 packed weights)
    float* ws     = (float*)d_ws;
    float* deg    = ws;                               // N      (becomes dinv)
    float* norm   = deg  + N;                         // E
    float* Hbuf   = norm + E;                         // N*HID
    float* AGG0   = Hbuf + (size_t)N * HID;           // N*HID  (becomes h)
    float* AGG1   = AGG0 + (size_t)N * HID;           // N*HID
    float* gsum   = AGG1 + (size_t)N * HID;           // G*HID
    float* gcnt   = gsum + NGRAPH * HID;              // G
    float* hidden = gcnt + NGRAPH;                    // G*MLPH
    _Float16* Bi0 = (_Float16*)(hidden + NGRAPH * MLPH);   // 256*128 halves
    _Float16* Br0 = Bi0 + 256 * HID;
    _Float16* Bi1 = Br0 + 256 * HID;                       // 128*128 halves
    _Float16* Br1 = Bi1 + 128 * HID;

    const long nodeElems = (long)N * HID;

    // 1. zero accumulators
    zero_kernel<<<(N + 255) / 256, 256, 0, stream>>>(deg, N);
    zero_kernel<<<(NGRAPH * HID + NGRAPH + 255) / 256, 256, 0, stream>>>(
        gsum, NGRAPH * HID + NGRAPH);

    // 2. pack weights to f16 WMMA-B layout
    pack_w_kernel<<<(256 * HID + 255) / 256, 256, 0, stream>>>(wi0, Bi0, 256);
    pack_w_kernel<<<(256 * HID + 255) / 256, 256, 0, stream>>>(wr0, Br0, 256);
    pack_w_kernel<<<(128 * HID + 255) / 256, 256, 0, stream>>>(wi1, Bi1, 128);
    pack_w_kernel<<<(128 * HID + 255) / 256, 256, 0, stream>>>(wr1, Br1, 128);

    // 3. gcn_norm
    deg_kernel <<<(E + 255) / 256, 256, 0, stream>>>(col, eattr, deg, E);
    dinv_kernel<<<(N + 255) / 256, 256, 0, stream>>>(deg, N);
    norm_kernel<<<(E + 255) / 256, 256, 0, stream>>>(row, col, eattr, deg, norm, E);

    // 4. conv0 (K = 256): 32-row strips x 8 column-tile waves
    gemm_dual_wmma<<<dim3(N / 32), dim3(32, 8), 0, stream>>>(
        x, Bi0, Br0, b0, Hbuf, AGG0, 256);
    scatter_kernel<<<(E + 3) / 4, 128, 0, stream>>>(row, col, norm, Hbuf, AGG0, E);
    relu_kernel<<<(nodeElems + 255) / 256, 256, 0, stream>>>(AGG0, nodeElems);

    // 5. conv1 (K = 128)
    gemm_dual_wmma<<<dim3(N / 32), dim3(32, 8), 0, stream>>>(
        AGG0, Bi1, Br1, b1, Hbuf, AGG1, 128);
    scatter_kernel<<<(E + 3) / 4, 128, 0, stream>>>(row, col, norm, Hbuf, AGG1, E);

    // 6. mean pool (relu fused into read)
    pool_kernel<<<((long)N * HID + 255) / 256, 256, 0, stream>>>(
        AGG1, batch, gsum, gcnt, N);

    // 7. MLP head
    mlp_hidden_kernel<<<(NGRAPH * MLPH + 255) / 256, 256, 0, stream>>>(
        gsum, gcnt, mw1, mb1, hidden);
    mlp_out_kernel<<<(NGRAPH * NCLS + 255) / 256, 256, 0, stream>>>(
        hidden, mw2, mb2, out);
}
